// Transolver_block_16836271800870
// MI455X (gfx1250) — compile-verified, matrix operand order checked by reading, not run on hardware
//
#include <hip/hip_runtime.h>
#include <hip/hip_bf16.h>
#include <cstdint>

#define DEV __device__ __forceinline__

typedef __attribute__((ext_vector_type(16))) __bf16 bf16x16;
typedef __attribute__((ext_vector_type(8)))  float  f32x8;
typedef __attribute__((ext_vector_type(4)))  unsigned int u32x4;
typedef __attribute__((ext_vector_type(8)))  int i32x8;
typedef __attribute__((ext_vector_type(4)))  int i32x4;

#if __has_builtin(__builtin_amdgcn_tensor_load_to_lds) && __has_builtin(__builtin_amdgcn_s_wait_tensorcnt)
#define USE_TDM 1
#if __clang_major__ >= 23
#define TDM_LOAD(g0, g1) __builtin_amdgcn_tensor_load_to_lds((g0), (g1), (i32x4){0,0,0,0}, (i32x4){0,0,0,0}, (i32x8){0,0,0,0,0,0,0,0}, 0)
#else
#define TDM_LOAD(g0, g1) __builtin_amdgcn_tensor_load_to_lds((g0), (g1), (i32x4){0,0,0,0}, (i32x4){0,0,0,0}, 0)
#endif
#else
#define USE_TDM 0
#endif

// ---------- helpers ----------
DEV uint16_t f2bf(float f) {
    uint32_t u = __builtin_bit_cast(uint32_t, f);
    uint32_t r = u + 0x7FFFu + ((u >> 16) & 1u);   // round-to-nearest-even
    return (uint16_t)(r >> 16);
}
DEV float gelu_f(float x) { return 0.5f * x * (1.0f + erff(x * 0.70710678118654752f)); }

DEV f32x8 wmma_bf16(bf16x16 a, bf16x16 b, f32x8 c) {
    return __builtin_amdgcn_wmma_f32_16x16x32_bf16(false, a, false, b, (short)0, c, false, false);
}

// Load one 16x32 bf16 A/B fragment per CDNA5 VGPR layout.
// p -> [row0][k0] of a [row][k] tile with row stride `rstride` (elements).
union FragU { bf16x16 v; uint32_t u[8]; };
DEV bf16x16 ld_frag(const uint16_t* p, int rstride) {
    const int lane = threadIdx.x & 31;
    const int half = lane >> 4;
    const int row  = lane & 15;
    FragU f;
#pragma unroll
    for (int j = 0; j < 8; ++j) {
        int kk = ((j < 4) ? (2 * j) : (8 + 2 * j)) + 8 * half;
        f.u[j] = *(const uint32_t*)(p + (size_t)row * rstride + kk);
    }
    return f.v;
}

#if USE_TDM
// D# group 0: count=1, lds_addr, 57-bit global byte address, type=2 ("image")
DEV u32x4 tdm_g0(uint32_t lds_off, const void* gptr) {
    uint64_t ga = (uint64_t)(uintptr_t)gptr;
    u32x4 g;
    g.x = 1u;
    g.y = lds_off;
    g.z = (uint32_t)ga;
    g.w = (uint32_t)((ga >> 32) & 0x01FFFFFFu) | (2u << 30);
    return g;
}
// D# group 1: data_size=1 (2 bytes), 2-D tile tile0 x tile1, tensor dim0/dim1, stride0.
DEV i32x8 tdm_g1(uint32_t dim0, uint32_t dim1, uint32_t tile0, uint32_t tile1,
                 uint64_t stride0) {
    i32x8 g;
    g[0] = (int)(1u << 16);                                      // data_size=1 (2B)
    g[1] = (int)((dim0 & 0xFFFFu) << 16);                        // tensor_dim0 lo
    g[2] = (int)((dim0 >> 16) | ((dim1 & 0xFFFFu) << 16));       // dim0 hi | dim1 lo
    g[3] = (int)((dim1 >> 16) | (tile0 << 16));                  // dim1 hi | tile_dim0
    g[4] = (int)(tile1 & 0xFFFFu);                               // tile_dim1, tile_dim2=0
    g[5] = (int)(uint32_t)stride0;                               // dim0_stride lo32
    g[6] = (int)(uint32_t)((stride0 >> 32) & 0xFFFFu);           // dim0_stride hi16
    g[7] = 0;
    return g;
}
#endif

// ---------- small utility kernels ----------
__global__ __launch_bounds__(256) void k_f2b(const float* __restrict__ s,
                                             uint16_t* __restrict__ d, int n) {
    int i = blockIdx.x * 256 + threadIdx.x;
    if (i < n) d[i] = f2bf(s[i]);
}
// transpose-convert: src [K][N] fp32 -> dst [N][K] bf16
__global__ __launch_bounds__(256) void k_f2bT(const float* __restrict__ s,
                                              uint16_t* __restrict__ d, int K, int N) {
    int i = blockIdx.x * 256 + threadIdx.x;
    if (i < K * N) {
        int k = i / N, n = i - k * N;
        d[(size_t)n * K + k] = f2bf(s[i]);
    }
}
__global__ __launch_bounds__(256) void k_zero(float* __restrict__ p, int n) {
    int i = blockIdx.x * 256 + threadIdx.x;
    if (i < n) p[i] = 0.0f;
}

// ---------- LayerNorm (D=512), fp32 in -> bf16 out ----------
__global__ __launch_bounds__(256) void k_ln(const float* __restrict__ x,
                                            const float* __restrict__ g,
                                            const float* __restrict__ b,
                                            uint16_t* __restrict__ out) {
    __shared__ float red[256];
    const int row = blockIdx.x, t = threadIdx.x;
    const float* xr = x + (size_t)row * 512;
    float e0 = xr[t], e1 = xr[t + 256];
    red[t] = e0 + e1; __syncthreads();
    for (int o = 128; o > 0; o >>= 1) { if (t < o) red[t] += red[t + o]; __syncthreads(); }
    float mean = red[0] * (1.0f / 512.0f);
    __syncthreads();
    float d0 = e0 - mean, d1 = e1 - mean;
    red[t] = d0 * d0 + d1 * d1; __syncthreads();
    for (int o = 128; o > 0; o >>= 1) { if (t < o) red[t] += red[t + o]; __syncthreads(); }
    float rstd = rsqrtf(red[0] * (1.0f / 512.0f) + 1e-5f);
    uint16_t* orow = out + (size_t)row * 512;
    orow[t]       = f2bf(d0 * rstd * g[t]       + b[t]);
    orow[t + 256] = f2bf(d1 * rstd * g[t + 256] + b[t + 256]);
}

// ---------- generic tiled bf16 GEMM, BM=128 BN=128 BK=32, 8 waves ----------
// A: [M][K] bf16 row-major.  Bt: [N][K] bf16 (pre-transposed weights).
// TDM double-buffered tile pipeline: wave0 DMAs tile t+1 while all waves WMMA tile t.
// MODE 0: xm scatter  -> outB[((b*8+h)*16384+n)*64+dh] = bf16(acc+bias)
// MODE 1: residual    -> outF[m*N+c] = resid[m*N+c] + acc + bias[c]
// MODE 2: gelu bf16   -> outB[m*N+c] = bf16(gelu(acc+bias[c]))
template <int MODE>
__global__ __launch_bounds__(256) void k_gemm(const uint16_t* __restrict__ A,
                                              const uint16_t* __restrict__ Bt,
                                              const float* __restrict__ bias,
                                              const float* __restrict__ resid,
                                              float* __restrict__ outF,
                                              uint16_t* __restrict__ outB,
                                              int M, int N, int K) {
    constexpr int BM = 128, BN = 128, BK = 32;
    __shared__ __align__(16) uint16_t lA[2][BM * BK];
    __shared__ __align__(16) uint16_t lB[2][BN * BK];
    const int tid = threadIdx.x;
    const int m0 = blockIdx.y * BM, n0 = blockIdx.x * BN;
    const int wid = tid >> 5, lane = tid & 31;
    const int wm = (wid & 3) * 32, wn = (wid >> 2) * 64;
    f32x8 acc[2][4] = {};
    const int nsteps = K / BK;

#if USE_TDM
    const uint32_t lA0 = (uint32_t)(uintptr_t)&lA[0][0];
    const uint32_t lB0 = (uint32_t)(uintptr_t)&lB[0][0];
    if (tid < 32) {   // wave 0 drives the tensor data mover
        TDM_LOAD(tdm_g0(lA0, A + (size_t)m0 * K), tdm_g1(K, M, BK, BM, K));
        TDM_LOAD(tdm_g0(lB0, Bt + (size_t)n0 * K), tdm_g1(K, N, BK, BN, K));
    }
    for (int t = 0; t < nsteps; ++t) {
        const int cur = t & 1;
        if (tid < 32) {
            if (t + 1 < nsteps) {
                const int kn = (t + 1) * BK;
                const uint32_t bo = (uint32_t)((cur ^ 1) * BM * BK * 2);
                TDM_LOAD(tdm_g0(lA0 + bo, A + (size_t)m0 * K + kn),
                         tdm_g1(K, M, BK, BM, K));
                TDM_LOAD(tdm_g0(lB0 + bo, Bt + (size_t)n0 * K + kn),
                         tdm_g1(K, N, BK, BN, K));
                __builtin_amdgcn_s_wait_tensorcnt(2);   // tile t landed (in-order)
            } else {
                __builtin_amdgcn_s_wait_tensorcnt(0);
            }
        }
        __syncthreads();
        bf16x16 fa[2], fb[4];
#pragma unroll
        for (int i = 0; i < 2; ++i) fa[i] = ld_frag(&lA[cur][(wm + i * 16) * BK], BK);
#pragma unroll
        for (int j = 0; j < 4; ++j) fb[j] = ld_frag(&lB[cur][(wn + j * 16) * BK], BK);
#pragma unroll
        for (int i = 0; i < 2; ++i)
#pragma unroll
            for (int j = 0; j < 4; ++j)
                acc[i][j] = wmma_bf16(fa[i], fb[j], acc[i][j]);
        __syncthreads();   // tile t fully consumed before TDM overwrites buffer
    }
#else
    for (int t = 0; t < nsteps; ++t) {
        const int k0 = t * BK;
        __syncthreads();
#pragma unroll
        for (int i = tid; i < (BM * BK) / 8; i += 256) {
            int e = i * 8, row = e >> 5, kk = e & 31;
            *(uint4*)&lA[0][e] = *(const uint4*)&A[(size_t)(m0 + row) * K + k0 + kk];
        }
#pragma unroll
        for (int i = tid; i < (BN * BK) / 8; i += 256) {
            int e = i * 8, col = e >> 5, kk = e & 31;
            *(uint4*)&lB[0][e] = *(const uint4*)&Bt[(size_t)(n0 + col) * K + k0 + kk];
        }
        if (k0 + BK < K) {
            __builtin_prefetch(&A[(size_t)(m0 + (tid >> 1)) * K + k0 + BK], 0, 1);
            __builtin_prefetch(&Bt[(size_t)(n0 + (tid >> 1)) * K + k0 + BK], 0, 1);
        }
        __syncthreads();
        bf16x16 fa[2], fb[4];
#pragma unroll
        for (int i = 0; i < 2; ++i) fa[i] = ld_frag(&lA[0][(wm + i * 16) * BK], BK);
#pragma unroll
        for (int j = 0; j < 4; ++j) fb[j] = ld_frag(&lB[0][(wn + j * 16) * BK], BK);
#pragma unroll
        for (int i = 0; i < 2; ++i)
#pragma unroll
            for (int j = 0; j < 4; ++j)
                acc[i][j] = wmma_bf16(fa[i], fb[j], acc[i][j]);
    }
#endif

    const int half = lane >> 4, ln = lane & 15;
#pragma unroll
    for (int i = 0; i < 2; ++i)
#pragma unroll
        for (int j = 0; j < 4; ++j)
#pragma unroll
            for (int r = 0; r < 8; ++r) {
                int row = m0 + wm + i * 16 + r + 8 * half;
                int col = n0 + wn + j * 16 + ln;
                float v = acc[i][j][r] + bias[col];
                if constexpr (MODE == 0) {
                    int b = row >> 14, n = row & 16383, h = col >> 6, dh = col & 63;
                    outB[((((size_t)b * 8 + h) << 14) + n) * 64 + dh] = f2bf(v);
                } else if constexpr (MODE == 1) {
                    size_t idx = (size_t)row * N + col;
                    outF[idx] = resid[idx] + v;
                } else {
                    outB[(size_t)row * N + col] = f2bf(gelu_f(v));
                }
            }
}

// ---------- slice assignment: temp MLP + gumbel softmax (per b,h; 128 pts/block) ----------
// wt1T / wsT are pre-transposed [G][Dh] bf16: B-fragments load straight from global.
__global__ __launch_bounds__(256) void k_slice(const uint16_t* __restrict__ xm,
                                               const uint16_t* __restrict__ wt1T,
                                               const uint16_t* __restrict__ wsT,
                                               const float* __restrict__ bt1,
                                               const float* __restrict__ wt2,
                                               const float* __restrict__ bt2,
                                               const float* __restrict__ biasH,
                                               const float* __restrict__ bs,
                                               const float* __restrict__ gum,
                                               uint16_t* __restrict__ sw,
                                               float* __restrict__ norm) {
    __shared__ float lT[128 * 64];
    __shared__ float lTemp[128];
    __shared__ float lw2[64], lb1[64], lbs[64], lNorm[64];
    const int tid = threadIdx.x;
    const int bh  = blockIdx.x, h = bh & 7;
    const int n0  = blockIdx.y * 128;
    if (tid < 64) { lw2[tid] = wt2[tid]; lb1[tid] = bt1[tid]; lbs[tid] = bs[tid]; lNorm[tid] = 0.0f; }
    __syncthreads();
    const int wid = tid >> 5, lane = tid & 31, half = lane >> 4, ln = lane & 15;
    const uint16_t* Abase = xm + ((size_t)bh * 16384 + n0 + wid * 16) * 64;
    f32x8 accT[4] = {}, accU[4] = {};
#pragma unroll
    for (int ks = 0; ks < 2; ++ks) {
        bf16x16 fa = ld_frag(Abase + ks * 32, 64);
#pragma unroll
        for (int j = 0; j < 4; ++j) {
            bf16x16 fb1 = ld_frag(wt1T + (j * 16) * 64 + ks * 32, 64);
            accT[j] = wmma_bf16(fa, fb1, accT[j]);
            bf16x16 fb2 = ld_frag(wsT + (j * 16) * 64 + ks * 32, 64);
            accU[j] = wmma_bf16(fa, fb2, accU[j]);
        }
    }
#pragma unroll
    for (int j = 0; j < 4; ++j)
#pragma unroll
        for (int r = 0; r < 8; ++r) {
            int lrow = wid * 16 + r + 8 * half, col = j * 16 + ln;
            lT[lrow * 64 + col] = accT[j][r] + lb1[col];
        }
    __syncthreads();
    const float bt2v = bt2[0], bhb = biasH[h];
    if (tid < 128) {
        float s = 0.f;
        for (int gg = 0; gg < 64; ++gg) s += gelu_f(lT[tid * 64 + gg]) * lw2[gg];
        lTemp[tid] = fmaxf(gelu_f(s + bt2v) + bhb, 0.01f);
    }
    __syncthreads();
#pragma unroll
    for (int j = 0; j < 4; ++j)
#pragma unroll
        for (int r = 0; r < 8; ++r) {
            int lrow = wid * 16 + r + 8 * half, col = j * 16 + ln;
            lT[lrow * 64 + col] = accU[j][r] + lbs[col];
        }
    __syncthreads();
    if (tid < 128) {
        const size_t rowg = (size_t)bh * 16384 + n0 + tid;
        const float* gp = gum + rowg * 64;
        const float invT = 1.0f / lTemp[tid];
        float mx = -1e30f;
        for (int gg = 0; gg < 64; ++gg) {
            float u = gp[gg];
            float gn = -logf(-logf(u + 1e-8f) + 1e-8f);
            float z  = (lT[tid * 64 + gg] + gn) * invT;
            lT[tid * 64 + gg] = z;
            mx = fmaxf(mx, z);
        }
        float ssum = 0.f;
        for (int gg = 0; gg < 64; ++gg) {
            float e = expf(lT[tid * 64 + gg] - mx);
            lT[tid * 64 + gg] = e; ssum += e;
        }
        const float inv = 1.0f / ssum;
        uint16_t* swr = sw + rowg * 64;
        for (int gg = 0; gg < 64; ++gg) {
            float p = lT[tid * 64 + gg] * inv;
            swr[gg] = f2bf(p);
            atomicAdd(&lNorm[gg], p);
        }
    }
    __syncthreads();
    if (tid < 64) atomicAdd(&norm[bh * 64 + tid], lNorm[tid]);
}

// ---------- slice tokens: st[g,c] = sum_n sw[n,g]*xm[n,c] / (norm+1e-5) ----------
__global__ __launch_bounds__(256) void k_tokens(const uint16_t* __restrict__ sw,
                                                const uint16_t* __restrict__ xm,
                                                const float* __restrict__ norm,
                                                float* __restrict__ st) {
    __shared__ __align__(16) uint16_t lA[64 * 32];   // sw^T [g][n]
    __shared__ __align__(16) uint16_t lB[64 * 32];   // xm^T [c][n]
    const int tid = threadIdx.x, bh = blockIdx.x;
    const int wid = tid >> 5, lane = tid & 31, half = lane >> 4, ln = lane & 15;
    const int wm = (wid & 3) * 16, wn = (wid >> 2) * 32;
    f32x8 acc[2] = {};
    const size_t base = (size_t)bh * 16384;
    for (int k0 = 0; k0 < 16384; k0 += 32) {
        __syncthreads();
#pragma unroll
        for (int i = tid; i < 2048; i += 256) {
            int col = i & 63, k = i >> 6;
            size_t gi = (base + k0 + k) * 64 + col;
            lA[col * 32 + k] = sw[gi];
            lB[col * 32 + k] = xm[gi];
        }
        __syncthreads();
        bf16x16 fa = ld_frag(&lA[wm * 32], 32);
#pragma unroll
        for (int j = 0; j < 2; ++j) {
            bf16x16 fb = ld_frag(&lB[(wn + j * 16) * 32], 32);
            acc[j] = wmma_bf16(fa, fb, acc[j]);
        }
    }
#pragma unroll
    for (int j = 0; j < 2; ++j)
#pragma unroll
        for (int r = 0; r < 8; ++r) {
            int g = wm + r + 8 * half, c = wn + j * 16 + ln;
            st[(size_t)bh * 4096 + g * 64 + c] = acc[j][r] / (norm[bh * 64 + g] + 1e-5f);
        }
}

// ---------- tiny 64x64 attention among slice tokens (VALU); emits ost^T [c][g] ----------
__global__ __launch_bounds__(256) void k_attn(const float* __restrict__ st,
                                              const float* __restrict__ wq,
                                              const float* __restrict__ wk,
                                              const float* __restrict__ wv,
                                              uint16_t* __restrict__ ostT) {
    __shared__ float bst[4096];
    __shared__ float bq[4096];
    __shared__ float bk2[4096];
    const int tid = threadIdx.x, bh = blockIdx.x;
    const float* stp = st + (size_t)bh * 4096;
    for (int i = tid; i < 4096; i += 256) bst[i] = stp[i];
    __syncthreads();
    for (int i = tid; i < 4096; i += 256) {
        int g = i >> 6, c = i & 63;
        float sq = 0.f, sk = 0.f;
        for (int d = 0; d < 64; ++d) {
            float sv = bst[g * 64 + d];
            sq += sv * wq[d * 64 + c];
            sk += sv * wk[d * 64 + c];
        }
        bq[i] = sq; bk2[i] = sk;
    }
    __syncthreads();
    float arow[64];
    if (tid < 64) {
        float mx = -1e30f;
        for (int j = 0; j < 64; ++j) {
            float s = 0.f;
            for (int d = 0; d < 64; ++d) s += bq[tid * 64 + d] * bk2[j * 64 + d];
            s *= 0.125f; arow[j] = s; mx = fmaxf(mx, s);
        }
        float ssum = 0.f;
        for (int j = 0; j < 64; ++j) { arow[j] = expf(arow[j] - mx); ssum += arow[j]; }
        float inv = 1.0f / ssum;
        for (int j = 0; j < 64; ++j) arow[j] *= inv;
    }
    __syncthreads();
    for (int i = tid; i < 4096; i += 256) {     // v overwrites q (dead)
        int g = i >> 6, c = i & 63;
        float sv = 0.f;
        for (int d = 0; d < 64; ++d) sv += bst[g * 64 + d] * wv[d * 64 + c];
        bq[i] = sv;
    }
    __syncthreads();
    if (tid < 64) {
        uint16_t* obase = ostT + (size_t)bh * 4096;
        for (int c = 0; c < 64; ++c) {
            float o = 0.f;
            for (int j = 0; j < 64; ++j) o += arow[j] * bq[j * 64 + c];
            obase[c * 64 + tid] = f2bf(o);      // transposed: [c][g]
        }
    }
}

// ---------- gather back: ctx[b,n,h*64+c] = sum_g sw[n,g]*ost[g,c] ----------
__global__ __launch_bounds__(256) void k_gather(const uint16_t* __restrict__ sw,
                                                const uint16_t* __restrict__ ostT,
                                                uint16_t* __restrict__ ctx) {
    const int tid = threadIdx.x, bh = blockIdx.x, n0 = blockIdx.y * 128;
    const int b = bh >> 3, h = bh & 7;
    const int wid = tid >> 5, lane = tid & 31, half = lane >> 4, ln = lane & 15;
    const uint16_t* Abase = sw + ((size_t)bh * 16384 + n0 + wid * 16) * 64;
    const uint16_t* Bbase = ostT + (size_t)bh * 4096;   // [c][g]
    f32x8 acc[4] = {};
#pragma unroll
    for (int ks = 0; ks < 2; ++ks) {
        bf16x16 fa = ld_frag(Abase + ks * 32, 64);
#pragma unroll
        for (int j = 0; j < 4; ++j) {
            bf16x16 fb = ld_frag(Bbase + (j * 16) * 64 + ks * 32, 64);
            acc[j] = wmma_bf16(fa, fb, acc[j]);
        }
    }
#pragma unroll
    for (int j = 0; j < 4; ++j)
#pragma unroll
        for (int r = 0; r < 8; ++r) {
            int n = n0 + wid * 16 + r + 8 * half, c = j * 16 + ln;
            ctx[((size_t)(b * 16384 + n)) * 512 + h * 64 + c] = f2bf(acc[j][r]);
        }
}

// ---------- host side ----------
extern "C" void kernel_launch(void* const* d_in, const int* in_sizes, int n_in,
                              void* d_out, int out_size, void* d_ws, size_t ws_size,
                              hipStream_t stream) {
    const float* fx    = (const float*)d_in[0];
    const float* gum   = (const float*)d_in[1];
    const float* ln1g  = (const float*)d_in[2];
    const float* ln1b  = (const float*)d_in[3];
    const float* wx    = (const float*)d_in[4];
    const float* bx    = (const float*)d_in[5];
    const float* wt1   = (const float*)d_in[6];
    const float* bt1   = (const float*)d_in[7];
    const float* wt2   = (const float*)d_in[8];
    const float* bt2   = (const float*)d_in[9];
    const float* biasH = (const float*)d_in[10];
    const float* ws_w  = (const float*)d_in[11];
    const float* bs    = (const float*)d_in[12];
    const float* wq    = (const float*)d_in[13];
    const float* wk    = (const float*)d_in[14];
    const float* wv    = (const float*)d_in[15];
    const float* wo    = (const float*)d_in[16];
    const float* bo    = (const float*)d_in[17];
    const float* ln2g  = (const float*)d_in[18];
    const float* ln2b  = (const float*)d_in[19];
    const float* w1    = (const float*)d_in[20];
    const float* b1    = (const float*)d_in[21];
    const float* w2    = (const float*)d_in[22];
    const float* b2    = (const float*)d_in[23];
    float* out = (float*)d_out;

    // workspace layout (bf16 stored as uint16_t); transposed weights are [N][K]
    const size_t EPD = 33554432ull;                 // B*N*D elements
    uint16_t* wsu  = (uint16_t*)d_ws;
    uint16_t* x1   = wsu;                           // [65536,512]
    uint16_t* xm   = wsu + EPD;                     // [B,H,N,64]
    uint16_t* swb  = wsu + 2 * EPD;                 // [B,H,N,64]
    uint16_t* ctx  = wsu + 3 * EPD;                 // [65536,512]
    uint16_t* hid  = wsu;                           // [65536,2048] reuses x1..ctx
    uint16_t* x2   = wsu + 4 * EPD;                 // [65536,512]
    uint16_t* wxT  = wsu + 5 * EPD;                 // [512,512]^T
    uint16_t* woT  = wxT + 262144;                  // [512,512]^T
    uint16_t* w1T  = woT + 262144;                  // [2048,512] (= w1^T)
    uint16_t* w2T  = w1T + 1048576;                 // [512,2048] (= w2^T)
    uint16_t* wt1T = w2T + 1048576;                 // [64,64]^T
    uint16_t* wsT  = wt1T + 4096;                   // [64,64]^T
    float*    norm = (float*)(wsT + 4096);          // [32,64]
    float*    st   = norm + 2048;                   // [32,64,64] fp32
    uint16_t* ostT = (uint16_t*)(st + 131072);      // [32,64,64] bf16 (c-major)

    // weight transpose-conversion + norm init
    k_f2bT<<<1024, 256, 0, stream>>>(wx, wxT, 512, 512);
    k_f2bT<<<1024, 256, 0, stream>>>(wo, woT, 512, 512);
    k_f2bT<<<4096, 256, 0, stream>>>(w1, w1T, 512, 2048);
    k_f2bT<<<4096, 256, 0, stream>>>(w2, w2T, 2048, 512);
    k_f2bT<<<16, 256, 0, stream>>>(wt1, wt1T, 64, 64);
    k_f2bT<<<16, 256, 0, stream>>>(ws_w, wsT, 64, 64);
    k_zero<<<8, 256, 0, stream>>>(norm, 2048);

    // attention branch
    k_ln<<<65536, 256, 0, stream>>>(fx, ln1g, ln1b, x1);
    k_gemm<0><<<dim3(4, 512), 256, 0, stream>>>(x1, wxT, bx, nullptr, nullptr, xm,
                                                65536, 512, 512);
    k_slice<<<dim3(32, 128), 256, 0, stream>>>(xm, wt1T, wsT, bt1, wt2, bt2, biasH,
                                               bs, gum, swb, norm);
    k_tokens<<<32, 256, 0, stream>>>(swb, xm, norm, st);
    k_attn<<<32, 256, 0, stream>>>(st, wq, wk, wv, ostT);
    k_gather<<<dim3(32, 128), 256, 0, stream>>>(swb, ostT, ctx);
    k_gemm<1><<<dim3(4, 512), 256, 0, stream>>>(ctx, woT, bo, fx, out, nullptr,
                                                65536, 512, 512);
    // MLP branch (in-place residual on d_out)
    k_ln<<<65536, 256, 0, stream>>>(out, ln2g, ln2b, x2);
    k_gemm<2><<<dim3(16, 512), 256, 0, stream>>>(x2, w1T, b1, nullptr, nullptr, hid,
                                                 65536, 2048, 512);
    k_gemm<1><<<dim3(4, 512), 256, 0, stream>>>(hid, w2T, b2, out, out, nullptr,
                                                65536, 512, 2048);
}
